// S6_18167711662500
// MI455X (gfx1250) — matrix operand order repaired and checked by reading, not compiled, not run on hardware
//
#include <hip/hip_runtime.h>
#include <hip/hip_bf16.h>

#define D_MODEL 1024
#define D_STATE 16
#define D_CONV  4
#define D_INNER 2048
#define BATCH   4
#define SEQ     2048
#define NROWS   (BATCH * SEQ)              // 8192
#define XP_N    (2 * D_STATE + D_INNER)    // 2080

typedef __attribute__((ext_vector_type(16))) __bf16 v16bf;
typedef __attribute__((ext_vector_type(8)))  __bf16 v8bf;
typedef __attribute__((ext_vector_type(8)))  float  v8f;

__device__ __forceinline__ float silu_f(float x) {
  return x * (1.0f / (1.0f + __expf(-x)));
}

// -----------------------------------------------------------------------------
// fp32 -> bf16 conversion (n even), coalesced float2 loads, packed b32 stores
// -----------------------------------------------------------------------------
__global__ __launch_bounds__(256) void cvt_f32_to_bf16(
    const float* __restrict__ in, __bf16* __restrict__ out, int n)
{
  const int i = (blockIdx.x * blockDim.x + threadIdx.x) * 2;
  if (i < n) {
    float2 f = *(const float2*)(in + i);
    out[i]     = (__bf16)f.x;
    out[i + 1] = (__bf16)f.y;
  }
}

// -----------------------------------------------------------------------------
// C[M,N] = A[M,K] * W[N,K]^T, A/W bf16 row-major, f32 accumulate.
// Block: 256 thr (8 waves), tile BM=128 x BN, K-chunk 32, LDS double-buffered.
// Every thread issues exactly 2 async-DMA ops per chunk (uniform ASYNCcnt):
//   A slice: 16 B (b128), B slice: BN/4 B (b128 for BN=64, b64 for BN=32).
// Pipeline: issue chunk k+1 -> s_wait_asynccnt 2 (chunk k complete) -> barrier
// -> WMMA from buf[k&1] -> barrier.
//   A-frag (ISA 7.12.2, 16-bit A 16x32): lane l -> row (l&15),
//     K runs [koff,koff+8) and [16+koff,+8), koff = 8*(l>=16)
//   B-frag: lane l -> col (l&15), same K runs as A-frag: [koff,koff+8) and [16+koff,+8)
//   C/D: vgpr r -> M = r + 8*(l>=16), N = l&15
// -----------------------------------------------------------------------------
template<int BN>
__global__ __launch_bounds__(256) void wmma_gemm_bf16(
    const __bf16* __restrict__ A, const __bf16* __restrict__ W,
    float* __restrict__ C, int M, int N, int K, int ldc)
{
  constexpr int BM = 128, KC = 32;
  constexpr int NT = BN / 16;                    // WMMA tiles per wave
  constexpr unsigned ABYTES = BM * KC * 2;       // 8 KB per buffer
  constexpr unsigned BBYTES = BN * KC * 2;       // 4/2 KB per buffer
  __shared__ __align__(16) __bf16 As[2][BM * KC];
  __shared__ __align__(16) __bf16 Bs[2][BN * KC];

  const int tid  = threadIdx.x;
  const int wave = tid >> 5;
  const int lane = tid & 31;
  const int nb   = N / BN;
  const int bm   = blockIdx.x / nb;
  const int bn   = blockIdx.x - bm * nb;

  const unsigned long long abase =
      (unsigned long long)(uintptr_t)(A + (size_t)bm * BM * K);
  const unsigned long long wbase =
      (unsigned long long)(uintptr_t)(W + (size_t)bn * BN * K);

  // A staging: chunk tid covers row = tid>>2, col-group = (tid&3)*8 (16 B)
  const unsigned ga_off = (unsigned)(((tid >> 2) * K + (tid & 3) * 8) * 2);
  // B staging: BN=64 -> 16 B chunks (row=tid>>2, cg=(tid&3)*8)
  //            BN=32 ->  8 B chunks (row=tid>>3, cg=(tid&7)*4)
  const unsigned gb_off = (BN == 64)
      ? (unsigned)(((tid >> 2) * K + (tid & 3) * 8) * 2)
      : (unsigned)(((tid >> 3) * K + (tid & 7) * 4) * 2);
  const unsigned lds_a0 = (unsigned)(uintptr_t)(&As[0][0]) + tid * 16;
  const unsigned lds_b0 = (unsigned)(uintptr_t)(&Bs[0][0]) + tid * (BN / 4);

  auto issue = [&](int k0, int ib) {
    const unsigned kb = (unsigned)(k0 * 2);
    asm volatile("global_load_async_to_lds_b128 %0, %1, %2"
                 :: "v"(lds_a0 + (unsigned)ib * ABYTES), "v"(ga_off + kb),
                    "s"(abase) : "memory");
    if constexpr (BN == 64) {
      asm volatile("global_load_async_to_lds_b128 %0, %1, %2"
                   :: "v"(lds_b0 + (unsigned)ib * BBYTES), "v"(gb_off + kb),
                      "s"(wbase) : "memory");
    } else {
      asm volatile("global_load_async_to_lds_b64 %0, %1, %2"
                   :: "v"(lds_b0 + (unsigned)ib * BBYTES), "v"(gb_off + kb),
                      "s"(wbase) : "memory");
    }
  };

  v8f acc[NT];
#pragma unroll
  for (int t = 0; t < NT; ++t) acc[t] = (v8f){0.f,0.f,0.f,0.f,0.f,0.f,0.f,0.f};

  const int fr   = lane & 15;
  const int koff = (lane >> 4) << 3;        // 0 or 8  (A-frag K split)
  const int bko  = (lane >> 4);             // chunk:  0 or 1 (B-frag K split)

  issue(0, 0);                               // prologue: stage chunk 0
  int ib = 0;
  for (int k0 = 0; k0 < K; k0 += KC, ib ^= 1) {
    if (k0 + KC < K) {
      issue(k0 + KC, ib ^ 1);                // stream next chunk behind compute
      asm volatile("s_wait_asynccnt 0x2" ::: "memory");   // chunk k0 complete
    } else {
      asm volatile("s_wait_asynccnt 0x0" ::: "memory");
    }
    __syncthreads();

    const v8bf* __restrict__ arow = (const v8bf*)(&As[ib][0] + (wave * 16 + fr) * KC);
    const v8bf alo = arow[koff >> 3];
    const v8bf ahi = arow[2 + (koff >> 3)];
    const v16bf afrag = __builtin_shufflevector(alo, ahi,
        0,1,2,3,4,5,6,7,8,9,10,11,12,13,14,15);

#pragma unroll
    for (int t = 0; t < NT; ++t) {
      const v8bf* __restrict__ brow = (const v8bf*)(&Bs[ib][0] + (t * 16 + fr) * KC);
      const v8bf blo = brow[bko];
      const v8bf bhi = brow[bko + 2];
      const v16bf bfrag = __builtin_shufflevector(blo, bhi,
          0,1,2,3,4,5,6,7,8,9,10,11,12,13,14,15);
      acc[t] = __builtin_amdgcn_wmma_f32_16x16x32_bf16(
          false, afrag, false, bfrag, (short)0, acc[t], false, false);
    }
    __syncthreads();                         // WAR: readers done before re-stage
  }

  const int m = bm * BM + wave * 16 + ((lane >> 4) << 3);
#pragma unroll
  for (int t = 0; t < NT; ++t) {
    const int n = bn * BN + t * 16 + fr;
    float* __restrict__ cp = C + (size_t)m * ldc + n;
#pragma unroll
    for (int r = 0; r < 8; ++r)
      cp[(size_t)r * ldc] = acc[t][r];
  }
}

// ---------------- depthwise causal conv1d (K=4) + SiLU, dual f32/bf16 out ---
__global__ __launch_bounds__(256) void conv_silu(
    const float* __restrict__ xz, const float* __restrict__ cw,
    const float* __restrict__ cb, float* __restrict__ xs,
    __bf16* __restrict__ xs_bf)
{
  const int idx = blockIdx.x * blockDim.x + threadIdx.x;   // NROWS*D_INNER
  const int d = idx & (D_INNER - 1);
  const int m = idx >> 11;                                  // row = b*SEQ + l
  const int l = m & (SEQ - 1);
  float acc = cb[d];
  const float* __restrict__ w = cw + d * D_CONV;
#pragma unroll
  for (int k = 0; k < D_CONV; ++k) {
    const int ll = l - (D_CONV - 1) + k;
    if (ll >= 0)
      acc = fmaf(w[k], xz[(size_t)(m - (D_CONV - 1) + k) * (2 * D_INNER) + d], acc);
  }
  const float s = silu_f(acc);
  xs[idx] = s;
  xs_bf[idx] = (__bf16)s;
}

// ---------------- dt = softplus(xp[:, :16] @ W_dt^T + b_dt) -----------------
__global__ __launch_bounds__(256) void dt_proj_softplus(
    const float* __restrict__ xp, const float* __restrict__ Wdt,
    const float* __restrict__ bdt, float* __restrict__ dt)
{
  const int idx = blockIdx.x * blockDim.x + threadIdx.x;   // NROWS*D_INNER
  const int d = idx & (D_INNER - 1);
  const int m = idx >> 11;
  const float* __restrict__ r = xp + (size_t)m * XP_N;      // first 16 cols
  const float* __restrict__ w = Wdt + d * D_STATE;
  float s = bdt[d];
#pragma unroll
  for (int n = 0; n < D_STATE; ++n) s = fmaf(r[n], w[n], s);
  dt[idx] = (s > 20.f) ? s : __logf(1.f + __expf(s));
}

// ---------------- selective scan: one thread per (b, d) ---------------------
__global__ __launch_bounds__(256) void selective_scan(
    const float* __restrict__ xs, const float* __restrict__ dt,
    const float* __restrict__ xp, const float* __restrict__ A_log,
    float* __restrict__ y)
{
  const int idx = blockIdx.x * blockDim.x + threadIdx.x;   // BATCH*D_INNER
  if (idx >= BATCH * D_INNER) return;
  const int d = idx & (D_INNER - 1);
  const int b = idx >> 11;

  float Ad[D_STATE];
#pragma unroll
  for (int n = 0; n < D_STATE; ++n) Ad[n] = -__expf(A_log[d * D_STATE + n]);
  float h[D_STATE];
#pragma unroll
  for (int n = 0; n < D_STATE; ++n) h[n] = 0.f;

  const size_t m0 = (size_t)b * SEQ;
  const float* __restrict__ xsrow = xs + m0 * D_INNER + d;
  const float* __restrict__ dtrow = dt + m0 * D_INNER + d;
  const float* __restrict__ xprow = xp + m0 * XP_N;
  float* __restrict__ yrow = y + m0 * D_INNER + d;

  for (int l = 0; l < SEQ; ++l) {
    const float dtv = dtrow[(size_t)l * D_INNER];
    const float xv  = xsrow[(size_t)l * D_INNER];
    const float* __restrict__ row = xprow + (size_t)l * XP_N;
    const float Cv  = row[2 * D_STATE + d];
    const float dtx = dtv * xv;
    if (l + 1 < SEQ) {   // global_prefetch_b8 for next timestep's strided loads
      __builtin_prefetch(dtrow + (size_t)(l + 1) * D_INNER, 0, 1);
      __builtin_prefetch(xsrow + (size_t)(l + 1) * D_INNER, 0, 1);
    }
    float ysum = 0.f;
#pragma unroll
    for (int n = 0; n < D_STATE; ++n) {
      const float bn = row[D_STATE + n];               // wave-uniform broadcast
      h[n] = fmaf(h[n], __expf(dtv * Ad[n]), dtx * bn);
      ysum += h[n];
    }
    yrow[(size_t)l * D_INNER] = ysum * Cv;
  }
}

// ---------------- gate: g = (y + xs*D) * silu(z), bf16 out ------------------
__global__ __launch_bounds__(256) void gate_skip(
    const float* __restrict__ y, const float* __restrict__ xs,
    const float* __restrict__ Dp, const float* __restrict__ xz,
    __bf16* __restrict__ g_bf)
{
  const int idx = blockIdx.x * blockDim.x + threadIdx.x;   // NROWS*D_INNER
  const int d = idx & (D_INNER - 1);
  const int m = idx >> 11;
  const float z = xz[(size_t)m * (2 * D_INNER) + D_INNER + d];
  const float v = fmaf(xs[idx], Dp[d], y[idx]);
  g_bf[idx] = (__bf16)(v * silu_f(z));
}

// -----------------------------------------------------------------------------
extern "C" void kernel_launch(void* const* d_in, const int* in_sizes, int n_in,
                              void* d_out, int out_size, void* d_ws, size_t ws_size,
                              hipStream_t stream)
{
  const float* x      = (const float*)d_in[0];
  const float* W_in   = (const float*)d_in[1];
  const float* conv_w = (const float*)d_in[2];
  const float* conv_b = (const float*)d_in[3];
  const float* W_x    = (const float*)d_in[4];
  const float* W_dt   = (const float*)d_in[5];
  const float* b_dt   = (const float*)d_in[6];
  const float* A_log  = (const float*)d_in[7];
  const float* Dp     = (const float*)d_in[8];
  const float* W_out  = (const float*)d_in[9];
  float* out = (float*)d_out;

  // ---- workspace layout (f32 arrays, then bf16 arrays) ----
  float* xz = (float*)d_ws;                            // [8192, 4096]
  float* xs = xz + (size_t)NROWS * 2 * D_INNER;        // [8192, 2048]
  float* xp = xs + (size_t)NROWS * D_INNER;            // [8192, 2080]
  float* dt = xp + (size_t)NROWS * XP_N;               // [8192, 2048]
  float* ys = dt + (size_t)NROWS * D_INNER;            // [8192, 2048]
  __bf16* x_bf   = (__bf16*)(ys + (size_t)NROWS * D_INNER);
  __bf16* xs_bf  = x_bf  + (size_t)NROWS * D_MODEL;
  __bf16* g_bf   = xs_bf + (size_t)NROWS * D_INNER;
  __bf16* Win_b  = g_bf  + (size_t)NROWS * D_INNER;
  __bf16* Wx_b   = Win_b + (size_t)(2 * D_INNER) * D_MODEL;
  __bf16* Wout_b = Wx_b  + (size_t)XP_N * D_INNER;

  const int TPB = 256;
  const int EW_BLOCKS = (NROWS * D_INNER) / TPB;       // 65536

  // 0) one-time bf16 operand conversions
  cvt_f32_to_bf16<<<(NROWS * D_MODEL) / (2 * TPB), TPB, 0, stream>>>(
      x, x_bf, NROWS * D_MODEL);
  cvt_f32_to_bf16<<<(2 * D_INNER * D_MODEL) / (2 * TPB), TPB, 0, stream>>>(
      W_in, Win_b, 2 * D_INNER * D_MODEL);
  cvt_f32_to_bf16<<<(XP_N * D_INNER) / (2 * TPB), TPB, 0, stream>>>(
      W_x, Wx_b, XP_N * D_INNER);
  cvt_f32_to_bf16<<<(D_MODEL * D_INNER) / (2 * TPB), TPB, 0, stream>>>(
      W_out, Wout_b, D_MODEL * D_INNER);

  // 1) xz = x @ W_in^T       [8192,4096] : 64 x 64 blocks
  wmma_gemm_bf16<64><<<(NROWS / 128) * (2 * D_INNER / 64), TPB, 0, stream>>>(
      x_bf, Win_b, xz, NROWS, 2 * D_INNER, D_MODEL, 2 * D_INNER);

  // 2) depthwise conv + SiLU (dual f32 + bf16 output)
  conv_silu<<<EW_BLOCKS, TPB, 0, stream>>>(xz, conv_w, conv_b, xs, xs_bf);

  // 3) xp = xs @ W_x^T       [8192,2080] : 64 x 65 blocks (BN=32)
  wmma_gemm_bf16<32><<<(NROWS / 128) * (XP_N / 32), TPB, 0, stream>>>(
      xs_bf, Wx_b, xp, NROWS, XP_N, D_INNER, XP_N);

  // 4) dt projection + softplus
  dt_proj_softplus<<<EW_BLOCKS, TPB, 0, stream>>>(xp, W_dt, b_dt, dt);

  // 5) selective scan (8192 lanes, sequential over L)
  selective_scan<<<(BATCH * D_INNER) / TPB, TPB, 0, stream>>>(xs, dt, xp, A_log, ys);

  // 6) skip + gating -> bf16
  gate_skip<<<EW_BLOCKS, TPB, 0, stream>>>(ys, xs, Dp, xz, g_bf);

  // 7) out = g @ W_out^T     [8192,1024] : 64 x 16 blocks
  wmma_gemm_bf16<64><<<(NROWS / 128) * (D_MODEL / 64), TPB, 0, stream>>>(
      g_bf, Wout_b, out, NROWS, D_MODEL, D_INNER, D_MODEL);
}
